// DecoderFactoredLSTMAtt_24859270709755
// MI455X (gfx1250) — compile-verified, matrix operand order checked by reading, not run on hardware
//
#include <hip/hip_runtime.h>
#include <hip/hip_bf16.h>

// Problem constants (match reference)
#define BB   64
#define TT   40
#define PP   49
#define FF   2048
#define EE   512
#define HH   1024
#define FAC  512
#define AA   512
#define VV   10000

typedef __attribute__((ext_vector_type(16))) __bf16 v16bf;
typedef __attribute__((ext_vector_type(8)))  __bf16 v8bf;
typedef __attribute__((ext_vector_type(4)))  __bf16 v4bf;
typedef __attribute__((ext_vector_type(8)))  float  v8f;

__device__ __forceinline__ __bf16 f2bf(float f) {
    union { float f; unsigned u; } a; a.f = f;
    unsigned u = a.u + 0x7FFFu + ((a.u >> 16) & 1u);   // round-to-nearest-even
    unsigned short h = (unsigned short)(u >> 16);
    union { unsigned short s; __bf16 b; } c; c.s = h;
    return c.b;
}

__device__ __forceinline__ float sigmoidf(float x) {
    return 1.f / (1.f + __expf(-x));
}

// ---------------------------------------------------------------------------
// Generic batched GEMM: C[b] = A[b] (MxK) @ W[b] (KxN) + bias[b] (N)
// bf16 WMMA, fp32 accumulate.
// Block = 256 threads = 8 waves (2 x 4), block tile 64x128, wave tile 32x32
// (2x2 fragments -> 4 v_wmma per K-chunk, A/B fragments each reused twice).
// Double-buffered LDS + register-staged global loads: ONE barrier per chunk.
// Preconditions (hold at every call site): M % 64 == 0, K % 32 == 0.
// N tail (V=10000) handled by a guarded non-pipelined path.
//
// LDS layout (per buffer):
//   lA  [M=64 ][K=32 (+8 pad)] row-major   -> A fragment = 2x ds_load_b128
//   lBt [N=128][K=32 (+8 pad)] transposed  -> B fragment = 2x ds_load_b128
// (CDNA5 16-bit WMMA layout: lane l needs K runs kh*8..+7 and 16+kh*8..+7,
//  contiguous in a [*][K] tile)
// ---------------------------------------------------------------------------
#define TILE_M 64
#define TILE_N 128
#define TILE_K 32
#define LDS_KS (TILE_K + 8)   // 40 halves = 80 bytes row stride (16B multiple)

__global__ __launch_bounds__(256)
void gemm_wmma_bf16(const float* __restrict__ A, const float* __restrict__ Wm,
                    const float* __restrict__ bias, float* __restrict__ C,
                    int M, int N, int K,
                    long long sA, long long sB, long long sBias, long long sC)
{
    const int batch = blockIdx.z;
    A    += (long long)batch * sA;
    Wm   += (long long)batch * sB;
    C    += (long long)batch * sC;
    bias += (long long)batch * sBias;

    __shared__ alignas(16) __bf16 lA [2][TILE_M][LDS_KS];
    __shared__ alignas(16) __bf16 lBt[2][TILE_N][LDS_KS];

    const int tid  = threadIdx.x;
    const int wid  = tid >> 5;       // wave 0..7
    const int lane = tid & 31;
    const int wm   = wid >> 2;       // 0..1 -> M offset 32*wm
    const int wn   = wid & 3;        // 0..3 -> N offset 32*wn

    const int m0 = blockIdx.y * TILE_M;
    const int n0 = blockIdx.x * TILE_N;
    const bool fullN = (n0 + TILE_N) <= N;

    v8f acc[2][2] = {};

    const int mfrag = lane & 15;     // row (A) / col (B) within 16
    const int khalf = lane >> 4;     // lanes 16..31 take the +8 K offset
    const int nchunks = K / TILE_K;

    // ---- staging helpers (inline lambdas, block-uniform control only) ----
    auto loadA = [&](int k0, float4 ra[2]) {
        #pragma unroll
        for (int i = 0; i < 2; ++i) {
            int idx = tid + i * 256;          // 0..511
            int r   = idx >> 3;               // 0..63
            int c4  = (idx & 7) * 4;          // 0,4,..,28
            ra[i] = *(const float4*)&A[(long long)(m0 + r) * K + k0 + c4];
        }
    };
    auto storeA = [&](const float4 ra[2], int buf) {
        #pragma unroll
        for (int i = 0; i < 2; ++i) {
            int idx = tid + i * 256;
            int r   = idx >> 3;
            int c4  = (idx & 7) * 4;
            v4bf p; p[0] = f2bf(ra[i].x); p[1] = f2bf(ra[i].y);
                    p[2] = f2bf(ra[i].z); p[3] = f2bf(ra[i].w);
            *(v4bf*)&lA[buf][r][c4] = p;
        }
    };
    auto loadB = [&](int k0, float4 rb[4]) {
        #pragma unroll
        for (int i = 0; i < 4; ++i) {
            int idx = tid + i * 256;          // 0..1023
            int r   = idx >> 5;               // K row 0..31
            int c4  = (idx & 31) * 4;         // N col 0,4,..,124
            rb[i] = *(const float4*)&Wm[(long long)(k0 + r) * N + n0 + c4];
        }
    };
    auto storeB = [&](const float4 rb[4], int buf) {
        #pragma unroll
        for (int i = 0; i < 4; ++i) {
            int idx = tid + i * 256;
            int r   = idx >> 5;
            int c4  = (idx & 31) * 4;
            lBt[buf][c4 + 0][r] = f2bf(rb[i].x);
            lBt[buf][c4 + 1][r] = f2bf(rb[i].y);
            lBt[buf][c4 + 2][r] = f2bf(rb[i].z);
            lBt[buf][c4 + 3][r] = f2bf(rb[i].w);
        }
    };
    auto stageB_tail = [&](int k0, int buf) {       // guarded scalar path
        #pragma unroll
        for (int i = 0; i < 16; ++i) {
            int idx = tid + i * 256;          // 0..4095
            int r   = idx >> 7;               // K row 0..31
            int cc  = idx & 127;              // N col 0..127
            float v = (n0 + cc < N) ? Wm[(long long)(k0 + r) * N + n0 + cc] : 0.f;
            lBt[buf][cc][r] = f2bf(v);
        }
    };
    auto compute = [&](int buf) {
        v16bf af[2], bf2[2];
        #pragma unroll
        for (int mi = 0; mi < 2; ++mi) {
            const v8bf* p0 = (const v8bf*)&lA[buf][wm * 32 + mi * 16 + mfrag][khalf * 8];
            const v8bf* p1 = (const v8bf*)&lA[buf][wm * 32 + mi * 16 + mfrag][khalf * 8 + 16];
            v8bf lo = p0[0], hi = p1[0];
            #pragma unroll
            for (int j = 0; j < 8; ++j) { af[mi][j] = lo[j]; af[mi][j + 8] = hi[j]; }
        }
        #pragma unroll
        for (int ni = 0; ni < 2; ++ni) {
            const v8bf* p0 = (const v8bf*)&lBt[buf][wn * 32 + ni * 16 + mfrag][khalf * 8];
            const v8bf* p1 = (const v8bf*)&lBt[buf][wn * 32 + ni * 16 + mfrag][khalf * 8 + 16];
            v8bf lo = p0[0], hi = p1[0];
            #pragma unroll
            for (int j = 0; j < 8; ++j) { bf2[ni][j] = lo[j]; bf2[ni][j + 8] = hi[j]; }
        }
        #pragma unroll
        for (int mi = 0; mi < 2; ++mi)
            #pragma unroll
            for (int ni = 0; ni < 2; ++ni)
                acc[mi][ni] = __builtin_amdgcn_wmma_f32_16x16x32_bf16(
                    false, af[mi], false, bf2[ni], (short)0, acc[mi][ni], false, false);
    };

    if (fullN) {
        // ---- software-pipelined, double-buffered main path ----
        float4 ra[2], rb[4];
        loadA(0, ra); loadB(0, rb);
        storeA(ra, 0); storeB(rb, 0);
        __syncthreads();
        for (int ck = 0; ck < nchunks; ++ck) {
            const int cur = ck & 1, nxt = cur ^ 1;
            const bool have_next = (ck + 1) < nchunks;
            if (have_next) {                    // issue next chunk's loads early
                loadA((ck + 1) * TILE_K, ra);
                loadB((ck + 1) * TILE_K, rb);
            }
            compute(cur);                       // WMMAs overlap in-flight loads
            if (have_next) {
                storeA(ra, nxt);
                storeB(rb, nxt);
            }
            __syncthreads();
        }
    } else {
        // ---- tail-N path (last vocab tile): simple, guarded staging ----
        float4 ra[2];
        for (int ck = 0; ck < nchunks; ++ck) {
            loadA(ck * TILE_K, ra);
            storeA(ra, 0);
            stageB_tail(ck * TILE_K, 0);
            __syncthreads();
            compute(0);
            __syncthreads();
        }
    }

    // C/D layout: VGPR r -> M = r + 8*(lane>=16); N = lane & 15
    const int ncol  = lane & 15;
    const int mbase = (lane >> 4) * 8;
    #pragma unroll
    for (int ni = 0; ni < 2; ++ni) {
        const int gn = n0 + wn * 32 + ni * 16 + ncol;
        if (gn >= N) continue;
        const float bi = bias[gn];
        #pragma unroll
        for (int mi = 0; mi < 2; ++mi) {
            #pragma unroll
            for (int r = 0; r < 8; ++r) {
                int gm = m0 + wm * 32 + mi * 16 + mbase + r;
                C[(long long)gm * N + gn] = acc[mi][ni][r] + bi;
            }
        }
    }
}

// ---------------------------------------------------------------------------
// Embedding gather: emb[b,t,e] = Wb[captions[b,t], e]
// ---------------------------------------------------------------------------
__global__ __launch_bounds__(256)
void embed_kernel(const int* __restrict__ captions, const float* __restrict__ Wb,
                  float* __restrict__ emb)
{
    long long idx = (long long)blockIdx.x * 256 + threadIdx.x;
    if (idx >= (long long)BB * TT * EE) return;
    int e = (int)(idx % EE);
    long long bt = idx / EE;
    int tok = captions[bt];
    emb[idx] = Wb[(long long)tok * EE + e];
}

// ---------------------------------------------------------------------------
// mean over P: mean_f[b,f] = (1/P) sum_p features[b,p,f]
// ---------------------------------------------------------------------------
__global__ __launch_bounds__(256)
void mean_kernel(const float* __restrict__ features, float* __restrict__ mean_f)
{
    int idx = blockIdx.x * 256 + threadIdx.x;
    if (idx >= BB * FF) return;
    int b = idx / FF, f = idx % FF;
    const float* fp = features + (long long)b * PP * FF + f;
    float s = 0.f;
    #pragma unroll 7
    for (int p = 0; p < PP; ++p) s += fp[(long long)p * FF];
    mean_f[idx] = s * (1.f / (float)PP);
}

// ---------------------------------------------------------------------------
// Attention: score -> softmax -> context (awe) + write alphas.  One block / b.
// ---------------------------------------------------------------------------
__global__ __launch_bounds__(256)
void attn_kernel(const float* __restrict__ att1,     // (B,P,A)
                 const float* __restrict__ att2,     // (B,A) incl. b_dec
                 const float* __restrict__ w_full,   // (A)
                 const float* __restrict__ b_full,   // (1)
                 const float* __restrict__ features, // (B,P,F)
                 float* __restrict__ awe,            // (B,F)
                 float* __restrict__ alphas_out,     // (B,T,P)
                 int t)
{
    const int b    = blockIdx.x;
    const int tid  = threadIdx.x;
    const int wid  = tid >> 5;
    const int lane = tid & 31;

    __shared__ float s_score[64];
    __shared__ float s_alpha[64];

    const float* a1 = att1 + (long long)b * PP * AA;
    const float* a2 = att2 + (long long)b * AA;

    for (int p = wid; p < PP; p += 8) {
        float sum = 0.f;
        for (int a = lane; a < AA; a += 32) {
            float v = a1[p * AA + a] + a2[a];
            v = v > 0.f ? v : 0.f;
            sum += v * w_full[a];
        }
        #pragma unroll
        for (int off = 16; off > 0; off >>= 1) sum += __shfl_xor(sum, off, 32);
        if (lane == 0) s_score[p] = sum + b_full[0];
    }
    __syncthreads();

    // softmax over P (redundant per-thread scan; P=49 is tiny)
    float mx = -1e30f;
    for (int p = 0; p < PP; ++p) mx = fmaxf(mx, s_score[p]);
    float den = 0.f;
    for (int p = 0; p < PP; ++p) den += __expf(s_score[p] - mx);
    if (tid < PP) {
        float al = __expf(s_score[tid] - mx) / den;
        s_alpha[tid] = al;
        alphas_out[((long long)b * TT + t) * PP + tid] = al;
    }
    __syncthreads();

    const float* fb = features + (long long)b * PP * FF;
    for (int f = tid; f < FF; f += 256) {
        float s = 0.f;
        #pragma unroll 7
        for (int p = 0; p < PP; ++p) s += s_alpha[p] * fb[(long long)p * FF + f];
        awe[(long long)b * FF + f] = s;
    }
}

// ---------------------------------------------------------------------------
// Build LSTM input: inp[b, :E] = emb[b,t,:]; inp[b, E:] = sigmoid(gpre)*awe
// ---------------------------------------------------------------------------
__global__ __launch_bounds__(256)
void build_inp_kernel(const float* __restrict__ emb,  // (B,T,E)
                      const float* __restrict__ gpre, // (B,F) incl. b_fbeta
                      const float* __restrict__ awe,  // (B,F)
                      float* __restrict__ inp, int t)
{
    int idx = blockIdx.x * 256 + threadIdx.x;
    if (idx >= BB * (EE + FF)) return;
    int b = idx / (EE + FF);
    int c = idx % (EE + FF);
    float v;
    if (c < EE) {
        v = emb[((long long)b * TT + t) * EE + c];
    } else {
        int f = c - EE;
        float g = sigmoidf(gpre[(long long)b * FF + f]);
        v = g * awe[(long long)b * FF + f];
    }
    inp[idx] = v;
}

// ---------------------------------------------------------------------------
// LSTM pointwise: gates i,f,o,c  -> update h, c, write hiddens[t]
// ---------------------------------------------------------------------------
__global__ __launch_bounds__(256)
void lstm_kernel(const float* __restrict__ u,  // (4,B,H) incl. bu
                 const float* __restrict__ w,  // (4,B,H) incl. bw
                 float* __restrict__ h, float* __restrict__ c,
                 float* __restrict__ hiddens, int t)
{
    int idx = blockIdx.x * 256 + threadIdx.x;
    if (idx >= BB * HH) return;
    const int S = BB * HH;
    float p0 = u[0 * S + idx] + w[0 * S + idx];
    float p1 = u[1 * S + idx] + w[1 * S + idx];
    float p2 = u[2 * S + idx] + w[2 * S + idx];
    float p3 = u[3 * S + idx] + w[3 * S + idx];
    float it = sigmoidf(p0);
    float ft = sigmoidf(p1);
    float ot = sigmoidf(p2);
    float ct = tanhf(p3);
    float cn = ft * c[idx] + it * ct;
    float hn = ot * cn;
    c[idx] = cn;
    h[idx] = hn;
    hiddens[(long long)t * S + idx] = hn;   // (T,B,H), idx = b*H + hcol
}

// ---------------------------------------------------------------------------
extern "C" void kernel_launch(void* const* d_in, const int* in_sizes, int n_in,
                              void* d_out, int out_size, void* d_ws, size_t ws_size,
                              hipStream_t stream)
{
    const int*   captions = (const int*)  d_in[0];
    // d_in[1] = lengths (unused, == T)
    const float* features = (const float*)d_in[2];
    const float* Wb       = (const float*)d_in[3];
    const float* w_init_h = (const float*)d_in[4];
    const float* b_init_h = (const float*)d_in[5];
    const float* w_init_c = (const float*)d_in[6];
    const float* b_init_c = (const float*)d_in[7];
    const float* w_enc    = (const float*)d_in[8];
    const float* b_enc    = (const float*)d_in[9];
    const float* w_dec    = (const float*)d_in[10];
    const float* b_dec    = (const float*)d_in[11];
    const float* w_full   = (const float*)d_in[12];
    const float* b_full   = (const float*)d_in[13];
    const float* w_fbeta  = (const float*)d_in[14];
    const float* b_fbeta  = (const float*)d_in[15];
    const float* Wv       = (const float*)d_in[16];
    const float* bv       = (const float*)d_in[17];
    const float* Ws       = (const float*)d_in[18];
    const float* bs       = (const float*)d_in[19];
    const float* Wu       = (const float*)d_in[20];
    const float* bu       = (const float*)d_in[21];
    const float* Ww       = (const float*)d_in[22];
    const float* bw       = (const float*)d_in[23];
    const float* Wc       = (const float*)d_in[24];
    const float* bc       = (const float*)d_in[25];

    float* out    = (float*)d_out;                         // (T*B, V)
    float* alphas = out + (long long)TT * BB * VV;         // (B, T, P)

    // Workspace carve-up (floats)
    float* ws = (float*)d_ws;
    long long off = 0;
    float* emb     = ws + off; off += (long long)BB * TT * EE;
    float* mean_f  = ws + off; off += (long long)BB * FF;
    float* hbuf    = ws + off; off += (long long)BB * HH;
    float* cbuf    = ws + off; off += (long long)BB * HH;
    float* att1    = ws + off; off += (long long)BB * PP * AA;
    float* att2    = ws + off; off += (long long)BB * AA;
    float* gpre    = ws + off; off += (long long)BB * FF;
    float* awe     = ws + off; off += (long long)BB * FF;
    float* inp     = ws + off; off += (long long)BB * (EE + FF);
    float* vbuf    = ws + off; off += 4LL * BB * FAC;
    float* sbuf    = ws + off; off += 4LL * BB * FAC;
    float* ubuf    = ws + off; off += 4LL * BB * HH;
    float* wbuf    = ws + off; off += 4LL * BB * HH;
    float* hiddens = ws + off; off += (long long)TT * BB * HH;
    (void)ws_size; (void)in_sizes; (void)n_in; (void)out_size;

    auto gemm = [&](const float* A, const float* Wm, const float* bias, float* C,
                    int M, int N, int K,
                    long long sA, long long sB, long long sBias, long long sC,
                    int batch) {
        dim3 grid((N + TILE_N - 1) / TILE_N, (M + TILE_M - 1) / TILE_M, batch);
        gemm_wmma_bf16<<<grid, 256, 0, stream>>>(A, Wm, bias, C, M, N, K,
                                                 sA, sB, sBias, sC);
    };

    // 1) Embedding gather
    {
        long long n = (long long)BB * TT * EE;
        embed_kernel<<<(int)((n + 255) / 256), 256, 0, stream>>>(captions, Wb, emb);
    }
    // 2) mean over P
    mean_kernel<<<(BB * FF + 255) / 256, 256, 0, stream>>>(features, mean_f);
    // 3) h0 / c0
    gemm(mean_f, w_init_h, b_init_h, hbuf, BB, HH, FF, 0, 0, 0, 0, 1);
    gemm(mean_f, w_init_c, b_init_c, cbuf, BB, HH, FF, 0, 0, 0, 0, 1);
    // 4) hoisted att1 = features @ w_enc + b_enc   (3136 x 2048 x 512)
    gemm(features, w_enc, b_enc, att1, BB * PP, AA, FF, 0, 0, 0, 0, 1);

    // 5) recurrence
    for (int t = 0; t < TT; ++t) {
        gemm(hbuf, w_dec, b_dec, att2, BB, AA, HH, 0, 0, 0, 0, 1);
        attn_kernel<<<BB, 256, 0, stream>>>(att1, att2, w_full, b_full,
                                            features, awe, alphas, t);
        gemm(hbuf, w_fbeta, b_fbeta, gpre, BB, FF, HH, 0, 0, 0, 0, 1);
        build_inp_kernel<<<(BB * (EE + FF) + 255) / 256, 256, 0, stream>>>(
            emb, gpre, awe, inp, t);
        gemm(inp, Wv, bv, vbuf, BB, FAC, EE + FF,
             0, (long long)(EE + FF) * FAC, FAC, (long long)BB * FAC, 4);
        gemm(vbuf, Ws, bs, sbuf, BB, FAC, FAC,
             (long long)BB * FAC, (long long)FAC * FAC, FAC, (long long)BB * FAC, 4);
        gemm(sbuf, Wu, bu, ubuf, BB, HH, FAC,
             (long long)BB * FAC, (long long)FAC * HH, HH, (long long)BB * HH, 4);
        gemm(hbuf, Ww, bw, wbuf, BB, HH, HH,
             0, (long long)HH * HH, HH, (long long)BB * HH, 4);
        lstm_kernel<<<(BB * HH + 255) / 256, 256, 0, stream>>>(
            ubuf, wbuf, hbuf, cbuf, hiddens, t);
    }

    // 6) vocab projection: outputs = hiddens (T*B x H) @ Wc + bc
    gemm(hiddens, Wc, bc, out, TT * BB, VV, HH, 0, 0, 0, 0, 1);
}
